// AttentionHead_18915035971785
// MI455X (gfx1250) — matrix operand order repaired and checked by reading, not compiled, not run on hardware
//
#include <hip/hip_runtime.h>
#include <math.h>

// ---------------------------------------------------------------------------
// Causal single-head attention for MI455X (gfx1250, wave32, WMMA).
//   B=8, T=2048, D=1024, HS=64.
// Phase 1: qkv_kernel  -> q,k f16 [B*T][64]; v f16 transposed [B][64][T]
// Phase 2: attn_kernel -> flash attention, v_wmma_f32_16x16x32_f16 for both
//          QK^T and P*V, online softmax, f32 output.
// WMMA fragment layouts used here (wave32; A/B use contiguous K, not the ISA table order):
//   A (16x32 f16): lane<16 holds row=lane, K {0..15}; lane>=16 K {16..31}
//   B (32x16 f16): lane holds col=lane&15, K {0..15} (lane<16) / {16..31} (lane>=16)
//   C/D (16x16 f32): VGPR r holds row r (lanes 0-15) / row 8+r (lanes 16-31), col=lane&15
// ---------------------------------------------------------------------------

typedef _Float16 half8  __attribute__((ext_vector_type(8)));
typedef _Float16 half16 __attribute__((ext_vector_type(16)));
typedef float    float8 __attribute__((ext_vector_type(8)));

#define WMMA_F16(a, b, c) \
    __builtin_amdgcn_wmma_f32_16x16x32_f16(false, (a), false, (b), (short)0, (c), false, false)

constexpr int Bn = 8, Tn = 2048, Dn = 1024, HSn = 64;
constexpr int ROWS = 128;   // token rows per workgroup
constexpr int KC   = 32;    // K-chunk
constexpr int NTOT = 192;   // q|k|v columns
constexpr int PADR = 40;    // padded row stride (halves): conflict-free b128 LDS reads

// B-fragment: 16 contiguous halves starting at p (two 16B loads).
__device__ inline half16 load_frag_b(const _Float16* p) {
    half8 a = *(const half8*)p;
    half8 b = *(const half8*)(p + 8);
    half16 r;
#pragma unroll
    for (int i = 0; i < 8; ++i) { r[i] = a[i]; r[i + 8] = b[i]; }
    return r;
}

// A-fragment: row pointer + lane-half select; contiguous K 16*hi..16*hi+15 (matches B).
__device__ inline half16 load_frag_a(const _Float16* rowp, int hi) {
    half8 lo = *(const half8*)(rowp + (hi ? 16 : 0));
    half8 h2 = *(const half8*)(rowp + (hi ? 24 : 8));
    half16 r;
#pragma unroll
    for (int i = 0; i < 8; ++i) { r[i] = lo[i]; r[i + 8] = h2[i]; }
    return r;
}

// ---------------------------------------------------------------------------
// Phase 1: fused QKV projection. grid = B*T/128 = 128 blocks, 256 threads.
// ---------------------------------------------------------------------------
__global__ __launch_bounds__(256) void qkv_kernel(
    const float* __restrict__ x,
    const float* __restrict__ Wq, const float* __restrict__ Wk, const float* __restrict__ Wv,
    _Float16* __restrict__ qh, _Float16* __restrict__ kh, _Float16* __restrict__ vt)
{
    __shared__ _Float16 xs[ROWS * PADR];   // x tile 128x32 (f16, padded)
    __shared__ _Float16 ws[NTOT * PADR];   // W chunk transposed: ws[n][k], 192x32

    const int tid  = threadIdx.x;
    const int wave = tid >> 5;
    const int lane = tid & 31;
    const int row  = lane & 15;
    const int hi   = lane >> 4;
    const int gRow0 = blockIdx.x * ROWS;   // global token row base within B*T

    float8 acc[12];
#pragma unroll
    for (int i = 0; i < 12; ++i) acc[i] = (float8)0.0f;

    for (int k0 = 0; k0 < Dn; k0 += KC) {
        // stage x tile: 1024 float4 reads, coalesced, f32->f16 into LDS
#pragma unroll
        for (int i = 0; i < 4; ++i) {
            int idx4 = i * 256 + tid;                 // 0..1023
            int r = idx4 >> 3, c = (idx4 & 7) * 4;
            float4 vx = *(const float4*)&x[(size_t)(gRow0 + r) * Dn + k0 + c];
            _Float16* d = &xs[r * PADR + c];
            d[0] = (_Float16)vx.x; d[1] = (_Float16)vx.y;
            d[2] = (_Float16)vx.z; d[3] = (_Float16)vx.w;
        }
        // stage W chunk transposed: 1536 float4 reads across Wq|Wk|Wv
#pragma unroll
        for (int i = 0; i < 6; ++i) {
            int idx4 = i * 256 + tid;                 // 0..1535
            int kk = idx4 / 48;
            int n  = (idx4 % 48) * 4;                 // 0..188
            const float* src = (n < 64)  ? &Wq[(size_t)(k0 + kk) * HSn + n]
                             : (n < 128) ? &Wk[(size_t)(k0 + kk) * HSn + (n - 64)]
                                         : &Wv[(size_t)(k0 + kk) * HSn + (n - 128)];
            float4 wv = *(const float4*)src;
            ws[(n + 0) * PADR + kk] = (_Float16)wv.x;
            ws[(n + 1) * PADR + kk] = (_Float16)wv.y;
            ws[(n + 2) * PADR + kk] = (_Float16)wv.z;
            ws[(n + 3) * PADR + kk] = (_Float16)wv.w;
        }
        __syncthreads();

        half16 afrag = load_frag_a(&xs[(wave * 16 + row) * PADR], hi);
#pragma unroll
        for (int nt = 0; nt < 12; ++nt) {
            half16 bfrag = load_frag_b(&ws[(nt * 16 + row) * PADR + (hi ? 16 : 0)]);
            acc[nt] = WMMA_F16(afrag, bfrag, acc[nt]);
        }
        __syncthreads();
    }

    // epilogue: q,k row-major f16; v transposed [B][64][T]
    const int b = gRow0 / Tn;
    const int tBase = gRow0 + wave * 16;
#pragma unroll
    for (int nt = 0; nt < 12; ++nt) {
        int n = (nt & 3) * 16 + row;
#pragma unroll
        for (int r = 0; r < 8; ++r) {
            int t = tBase + r + hi * 8;               // global token row
            _Float16 val = (_Float16)acc[nt][r];
            if (nt < 4)      qh[(size_t)t * HSn + n] = val;
            else if (nt < 8) kh[(size_t)t * HSn + n] = val;
            else             vt[((size_t)b * HSn + n) * Tn + (t - b * Tn)] = val;
        }
    }
}

// ---------------------------------------------------------------------------
// Phase 2: causal flash attention. grid = B * (T/128) = 128 blocks, 256 thr.
// Each wave owns 16 query rows; key loop in blocks of 32.
// ---------------------------------------------------------------------------
__global__ __launch_bounds__(256) void attn_kernel(
    const _Float16* __restrict__ qh, const _Float16* __restrict__ kh,
    const _Float16* __restrict__ vt, float* __restrict__ out)
{
    __shared__ _Float16 ps[8 * 16 * PADR];   // per-wave P scratch (16x32 + pad)

    const int tid  = threadIdx.x;
    const int wave = tid >> 5;
    const int lane = tid & 31;
    const int row  = lane & 15;
    const int hi   = lane >> 4;
    const int b     = blockIdx.x >> 4;                       // 16 q-blocks / batch
    const int qbase = (blockIdx.x & 15) * 128 + wave * 16;   // query base (in-batch)

    const _Float16* qb = qh + (size_t)b * Tn * HSn;
    const _Float16* kb = kh + (size_t)b * Tn * HSn;
    const _Float16* vb = vt + (size_t)b * HSn * Tn;

    // preload Q A-fragments (16 rows x 64 HS = two 16x32 frags)
    const _Float16* qrow = qb + (size_t)(qbase + row) * HSn;
    half16 qf0 = load_frag_a(qrow, hi);
    half16 qf1 = load_frag_a(qrow + 32, hi);

    float8 o[4];
#pragma unroll
    for (int t = 0; t < 4; ++t) o[t] = (float8)0.0f;
    float mrun[8], lrun[8];
#pragma unroll
    for (int r = 0; r < 8; ++r) { mrun[r] = -3.0e38f; lrun[r] = 0.0f; }

    _Float16* pw = &ps[wave * 16 * PADR];
    const float scale = 0.125f;                 // HS^-0.5
    const int kend = qbase + 16;                // causal bound (exclusive)

    for (int t0 = 0; t0 < kend; t0 += 32) {
        if (t0 + 32 < kend) {                   // global_prefetch_b8 next block
            __builtin_prefetch(kb + (size_t)(t0 + 32 + row) * HSn, 0, 1);
            __builtin_prefetch(vb + (size_t)row * Tn + t0 + 32, 0, 1);
        }

        // ---- S = Q K^T for 32 keys (two 16x16 tiles, K-dim = HS = 64)
        float8 s0 = (float8)0.0f, s1 = (float8)0.0f;
        const _Float16* kp0 = kb + (size_t)(t0 + row) * HSn + hi * 16;
        s0 = WMMA_F16(qf0, load_frag_b(kp0),      s0);
        s0 = WMMA_F16(qf1, load_frag_b(kp0 + 32), s0);
        const _Float16* kp1 = kp0 + (size_t)16 * HSn;
        s1 = WMMA_F16(qf0, load_frag_b(kp1),      s1);
        s1 = WMMA_F16(qf1, load_frag_b(kp1 + 32), s1);

        // ---- scale + causal mask + per-row block max
        float pmax[8];
#pragma unroll
        for (int r = 0; r < 8; ++r) {
            int qr = qbase + r + hi * 8;
            float v0 = (t0 + row      <= qr) ? s0[r] * scale : -3.0e38f;
            float v1 = (t0 + 16 + row <= qr) ? s1[r] * scale : -3.0e38f;
            s0[r] = v0; s1[r] = v1;
            pmax[r] = fmaxf(v0, v1);
        }
#pragma unroll
        for (int m = 1; m < 16; m <<= 1)
#pragma unroll
            for (int r = 0; r < 8; ++r)
                pmax[r] = fmaxf(pmax[r], __shfl_xor(pmax[r], m, 16));

        // ---- online softmax update
        float alpha[8], rsum[8];
#pragma unroll
        for (int r = 0; r < 8; ++r) {
            float mnew = fmaxf(mrun[r], pmax[r]);
            alpha[r] = __expf(mrun[r] - mnew);
            mrun[r] = mnew;
            float p0 = __expf(s0[r] - mnew);
            float p1 = __expf(s1[r] - mnew);
            s0[r] = p0; s1[r] = p1;
            rsum[r] = p0 + p1;
        }
#pragma unroll
        for (int m = 1; m < 16; m <<= 1)
#pragma unroll
            for (int r = 0; r < 8; ++r)
                rsum[r] += __shfl_xor(rsum[r], m, 16);
#pragma unroll
        for (int r = 0; r < 8; ++r) lrun[r] = lrun[r] * alpha[r] + rsum[r];
#pragma unroll
        for (int t = 0; t < 4; ++t)
#pragma unroll
            for (int r = 0; r < 8; ++r) o[t][r] *= alpha[r];

        // ---- relayout P (C-layout -> A-fragment) via per-wave LDS scratch
#pragma unroll
        for (int r = 0; r < 8; ++r) {
            int m = r + hi * 8;
            pw[m * PADR + row]      = (_Float16)s0[r];
            pw[m * PADR + 16 + row] = (_Float16)s1[r];
        }
        asm volatile("s_wait_dscnt 0" ::: "memory");
        half16 pf = load_frag_a(&pw[row * PADR], hi);

        // ---- O += P * V (V^T rows are contiguous in keys)
#pragma unroll
        for (int ht = 0; ht < 4; ++ht) {
            const _Float16* vp = vb + (size_t)(ht * 16 + row) * Tn + t0 + hi * 16;
            o[ht] = WMMA_F16(pf, load_frag_b(vp), o[ht]);
        }
    }

    // ---- epilogue: normalize, f32 store
    float* ob = out + ((size_t)b * Tn + qbase) * HSn;
#pragma unroll
    for (int ht = 0; ht < 4; ++ht)
#pragma unroll
        for (int r = 0; r < 8; ++r) {
            int m = r + hi * 8;
            ob[(size_t)m * HSn + ht * 16 + row] = o[ht][r] / lrun[r];
        }
}

// ---------------------------------------------------------------------------
extern "C" void kernel_launch(void* const* d_in, const int* in_sizes, int n_in,
                              void* d_out, int out_size, void* d_ws, size_t ws_size,
                              hipStream_t stream) {
    (void)in_sizes; (void)n_in; (void)out_size; (void)ws_size;
    const float* x  = (const float*)d_in[0];
    const float* Wq = (const float*)d_in[1];
    const float* Wk = (const float*)d_in[2];
    const float* Wv = (const float*)d_in[3];
    float* out = (float*)d_out;

    _Float16* qh = (_Float16*)d_ws;                       // [B*T][64]
    _Float16* kh = qh + (size_t)Bn * Tn * HSn;            // [B*T][64]
    _Float16* vt = kh + (size_t)Bn * Tn * HSn;            // [B][64][T]

    qkv_kernel<<<(Bn * Tn) / ROWS, 256, 0, stream>>>(x, Wq, Wk, Wv, qh, kh, vt);
    attn_kernel<<<Bn * (Tn / 128), 256, 0, stream>>>(qh, kh, vt, out);
}
